// ParakeetEncoder_12360915878712
// MI455X (gfx1250) — compile-verified, matrix operand order checked by reading, not run on hardware
//
#include <hip/hip_runtime.h>
#include <hip/hip_bf16.h>

typedef __bf16 bf16;
typedef __attribute__((ext_vector_type(8)))  bf16  v8bf;
typedef __attribute__((ext_vector_type(16))) bf16  v16bf;
typedef __attribute__((ext_vector_type(8)))  float v8f;

#define EPSV 1e-5f
#define BB   2
#define TT   512
#define HH   1024
#define II   4096
#define NHH  8
#define HDD  128
#define NL   8
#define LRELN 1023
#define K9   9
#define CCC  256
#define BT   (BB*TT)
#define LDT  40   // padded LDS row stride (bf16 elements): conflict-free b128 frags

// ---------------------------------------------------------------- async copy
// Generic __shared__ pointers carry the wave-relative LDS byte offset in the
// low 32 bits (flat->LDS truncates addr[31:0]), which is exactly the VDST
// operand GLOBAL_LOAD_ASYNC_TO_LDS wants.
__device__ __forceinline__ unsigned lds_off32(const void* p) {
  return (unsigned)(unsigned long long)p;
}
__device__ __forceinline__ void async_b128(unsigned lds_off, const void* g) {
  asm volatile("global_load_async_to_lds_b128 %0, %1, off"
               :: "v"(lds_off), "v"(g) : "memory");
}
__device__ __forceinline__ void wait_async0() {
#if __has_builtin(__builtin_amdgcn_s_wait_asynccnt)
  __builtin_amdgcn_s_wait_asynccnt(0);
#else
  asm volatile("s_wait_asynccnt 0x0" ::: "memory");
#endif
}

// ---------------------------------------------------------------- WMMA GEMM
__device__ __forceinline__ v16bf ldfrag(const bf16* p) {
  v8bf lo = *(const v8bf*)p;
  v8bf hi = *(const v8bf*)(p + 16);
  return __builtin_shufflevector(lo, hi, 0,1,2,3,4,5,6,7,8,9,10,11,12,13,14,15);
}
__device__ __forceinline__ int imin(int a, int b) { return a < b ? a : b; }

// C[M,N] = act(alpha * A[M,K] @ op(B) + bias) + add
// TRANSB=1: B is N x K (row-major, like weights).  TRANSB=0: B is K x N.
// Requires K % 32 == 0 (true for all call sites). Ragged M/N handled by
// clamped loads (masked at store): staging is branch-free.
// Block tile 64(M) x 128(N), 8 waves each computing 32x32; double-buffered LDS
// filled by async global->LDS DMA (ASYNCcnt), one barrier per K-tile.
template<int TRANSB>
__global__ __launch_bounds__(256)
void wmma_gemm(const bf16* __restrict__ A, const bf16* __restrict__ B,
               float* __restrict__ Cf, bf16* __restrict__ Cb,
               const float* __restrict__ bias, const float* __restrict__ add,
               int M, int N, int K, int lda, int ldb, int ldc,
               long long sAb, long long sAh, long long sBb, long long sBh,
               long long sCb, long long sCh, int nH,
               float alpha, int act)
{
  __shared__ __align__(16) bf16 As[2][64 * LDT];
  __shared__ __align__(16) bf16 Bs[2][128 * LDT];

  const int tid = threadIdx.x;
  const int z   = blockIdx.z;
  const int bb  = z / nH, hh = z % nH;
  const bf16* Aa = A + (long long)bb * sAb + (long long)hh * sAh;
  const bf16* Ba = B + (long long)bb * sBb + (long long)hh * sBh;
  const long long coff = (long long)bb * sCb + (long long)hh * sCh;

  const int m0 = blockIdx.y * 64;
  const int n0 = blockIdx.x * 128;

  const int lane  = tid & 31;
  const int w     = tid >> 5;
  const int wn    = w & 3;            // N sub-block 0..3 (32 cols each)
  const int wm    = w >> 2;           // 0..1 -> rows wm*32
  const int lrow  = lane & 15;
  const int khalf = (lane >> 4) << 3; // 0 or 8 (element offset in K)

  // staging indices (branch-free, row-clamped)
  const int ar = tid >> 2, ac = (tid & 3) << 3;   // A: 64x32, 8 elem/thread
  const int br = tid >> 1, bc = (tid & 1) << 4;   // B(NxK): 128x32, 16 elem/thread
  const int bk = tid >> 3, bn = (tid & 7) << 4;   // B(KxN): 32x128, 16 elem/thread
  const long long arow = (long long)imin(m0 + ar, M - 1) * lda;
  const long long brow = TRANSB ? (long long)imin(n0 + br, N - 1) * ldb : 0;

  const bf16* gA = Aa + arow + ac;                // this thread's A chunk, K-tile 0
  const bf16* gB = Ba + brow + bc;                // this thread's B chunk (TRANSB=1)
  const unsigned ldsA[2]  = { lds_off32(&As[0][ar * LDT + ac]),
                              lds_off32(&As[1][ar * LDT + ac]) };
  const unsigned ldsB[2]  = { lds_off32(&Bs[0][br * LDT + bc]),
                              lds_off32(&Bs[1][br * LDT + bc]) };

  bf16 b_s[16];

  // prologue: DMA K-tile 0 into buffer 0
  async_b128(ldsA[0], gA);
  if (TRANSB) {
    async_b128(ldsB[0],      gB);
    async_b128(ldsB[0] + 16, gB + 8);
  } else {
    for (int j = 0; j < 16; ++j)
      b_s[j] = Ba[(long long)bk * ldb + imin(n0 + bn + j, N - 1)];
    for (int j = 0; j < 16; ++j) Bs[0][(bn + j) * LDT + bk] = b_s[j];
  }
  wait_async0();
  __syncthreads();

  v8f c00 = {0,0,0,0,0,0,0,0}, c01 = {0,0,0,0,0,0,0,0};
  v8f c10 = {0,0,0,0,0,0,0,0}, c11 = {0,0,0,0,0,0,0,0};

  int cur = 0;
  for (int k0 = 0; k0 < K; k0 += 32) {
    const int k1 = k0 + 32;
    const bool more = k1 < K;
    const int nxt = cur ^ 1;
    if (more) {  // DMA next K-tile into the other buffer (overlaps WMMA below)
      async_b128(ldsA[nxt], gA + k1);
      if (TRANSB) {
        async_b128(ldsB[nxt],      gB + k1);
        async_b128(ldsB[nxt] + 16, gB + k1 + 8);
      } else {
        for (int j = 0; j < 16; ++j)
          b_s[j] = Ba[(long long)(k1 + bk) * ldb + imin(n0 + bn + j, N - 1)];
      }
    }

    const bf16* Ab = As[cur];
    const bf16* Bb = Bs[cur];
    v16bf a0 = ldfrag(&Ab[(wm * 32 +      lrow) * LDT + khalf]);
    v16bf a1 = ldfrag(&Ab[(wm * 32 + 16 + lrow) * LDT + khalf]);
    v16bf b0 = ldfrag(&Bb[(wn * 32 +      lrow) * LDT + khalf]);
    v16bf b1 = ldfrag(&Bb[(wn * 32 + 16 + lrow) * LDT + khalf]);
    c00 = __builtin_amdgcn_wmma_f32_16x16x32_bf16(false, a0, false, b0, (short)0, c00, false, false);
    c01 = __builtin_amdgcn_wmma_f32_16x16x32_bf16(false, a0, false, b1, (short)0, c01, false, false);
    c10 = __builtin_amdgcn_wmma_f32_16x16x32_bf16(false, a1, false, b0, (short)0, c10, false, false);
    c11 = __builtin_amdgcn_wmma_f32_16x16x32_bf16(false, a1, false, b1, (short)0, c11, false, false);

    if (more) {
      if (!TRANSB)
        for (int j = 0; j < 16; ++j) Bs[nxt][(bn + j) * LDT + bk] = b_s[j];
      wait_async0();   // our wave's DMA into buf[nxt] done before the barrier
    }
    __syncthreads();
    cur ^= 1;
  }

  // epilogue
  const int hi8 = (lane >> 4) << 3;
  for (int sm = 0; sm < 2; ++sm) {
    for (int sn = 0; sn < 2; ++sn) {
      v8f cc = sm ? (sn ? c11 : c10) : (sn ? c01 : c00);
      const int n  = n0 + wn * 32 + sn * 16 + lrow;
      const int mb = m0 + wm * 32 + sm * 16 + hi8;
      for (int r = 0; r < 8; ++r) {
        int m = mb + r;
        if (m < M && n < N) {
          float v = cc[r] * alpha;
          if (bias) v += bias[n];
          if (act == 1)      v = v / (1.f + __expf(-v));   // SiLU
          else if (act == 2) v = fmaxf(v, 0.f);            // ReLU
          long long ci = coff + (long long)m * ldc + n;
          if (add) v += add[ci];
          if (Cf) Cf[ci] = v;
          if (Cb) Cb[ci] = (bf16)v;
        }
      }
    }
  }
}

// ---------------------------------------------------------------- elementwise
__global__ void cast_f2b(const float* __restrict__ in, bf16* __restrict__ out, long long n) {
  long long i = (long long)blockIdx.x * blockDim.x + threadIdx.x;
  if (i < n) out[i] = (bf16)in[i];
}

__global__ __launch_bounds__(256)
void layernorm_k(const float* __restrict__ in, const float* __restrict__ g,
                 const float* __restrict__ b, float* __restrict__ outf,
                 bf16* __restrict__ outb)
{
  __shared__ float r0[256], r1[256];
  int row = blockIdx.x;
  const float* xr = in + (long long)row * HH;
  float v[4]; float sum = 0.f, sq = 0.f;
  for (int k = 0; k < 4; ++k) {
    int c = threadIdx.x + k * 256;
    v[k] = xr[c]; sum += v[k]; sq += v[k] * v[k];
  }
  r0[threadIdx.x] = sum; r1[threadIdx.x] = sq; __syncthreads();
  for (int st = 128; st > 0; st >>= 1) {
    if (threadIdx.x < st) { r0[threadIdx.x] += r0[threadIdx.x + st]; r1[threadIdx.x] += r1[threadIdx.x + st]; }
    __syncthreads();
  }
  float mean = r0[0] * (1.f / HH);
  float var  = r1[0] * (1.f / HH) - mean * mean;
  float rstd = rsqrtf(var + EPSV);
  for (int k = 0; k < 4; ++k) {
    int c = threadIdx.x + k * 256;
    float o = (v[k] - mean) * rstd * g[c] + b[c];
    if (outf) outf[(long long)row * HH + c] = o;
    if (outb) outb[(long long)row * HH + c] = (bf16)o;
  }
}

__global__ void q_bias_k(const float* __restrict__ q, const float* __restrict__ bu,
                         const float* __restrict__ bv, bf16* __restrict__ qu,
                         bf16* __restrict__ qv, long long n) {
  long long i = (long long)blockIdx.x * blockDim.x + threadIdx.x;
  if (i < n) {
    int c = (int)(i % HH);
    float qq = q[i];
    qu[i] = (bf16)(qq + bu[c]);
    qv[i] = (bf16)(qq + bv[c]);
  }
}

__global__ void pe_make_k(bf16* __restrict__ pe) {
  long long i = (long long)blockIdx.x * blockDim.x + threadIdx.x;
  long long n = (long long)LRELN * HH;
  if (i >= n) return;
  int d = (int)(i % HH); int l = (int)(i / HH);
  float p  = (float)(TT - 1 - l);
  int de   = d & ~1;
  float dv = __expf(-logf(10000.f) * (float)de / (float)HH);
  float a  = p * dv;
  pe[i] = (bf16)((d & 1) ? cosf(a) : sinf(a));
}

// fused rel_shift + scale + softmax -> bf16 probabilities
__global__ __launch_bounds__(256)
void softmax_rel_k(const float* __restrict__ scc, const float* __restrict__ scp,
                   bf16* __restrict__ P) {
  __shared__ float red[256];
  int t  = blockIdx.x;
  int bh = blockIdx.y;
  const float* rc = scc + ((long long)bh * TT + t) * TT;
  const float* rp = scp + ((long long)bh * TT + t) * LRELN;
  int shift = TT - 1 - t;
  const float scale = 0.08838834764831845f; // 1/sqrt(128)
  float x0 = (rc[threadIdx.x]       + rp[shift + threadIdx.x])       * scale;
  float x1 = (rc[threadIdx.x + 256] + rp[shift + threadIdx.x + 256]) * scale;
  red[threadIdx.x] = fmaxf(x0, x1); __syncthreads();
  for (int st = 128; st > 0; st >>= 1) {
    if (threadIdx.x < st) red[threadIdx.x] = fmaxf(red[threadIdx.x], red[threadIdx.x + st]);
    __syncthreads();
  }
  float mx = red[0]; __syncthreads();
  float e0 = __expf(x0 - mx), e1 = __expf(x1 - mx);
  red[threadIdx.x] = e0 + e1; __syncthreads();
  for (int st = 128; st > 0; st >>= 1) {
    if (threadIdx.x < st) red[threadIdx.x] += red[threadIdx.x + st];
    __syncthreads();
  }
  float inv = 1.f / red[0];
  bf16* pr = P + ((long long)bh * TT + t) * TT;
  pr[threadIdx.x]       = (bf16)(e0 * inv);
  pr[threadIdx.x + 256] = (bf16)(e1 * inv);
}

__global__ void glu_t_k(const float* __restrict__ y2, float* __restrict__ yt, long long n) {
  long long i = (long long)blockIdx.x * blockDim.x + threadIdx.x;
  if (i >= n) return;                 // i over (b,t,c)
  int c = (int)(i % HH);
  long long bt = i / HH;
  int t = (int)(bt % TT), b = (int)(bt / TT);
  const float* row = y2 + (long long)(b * TT + t) * (2 * HH);
  float a = row[c], gv = row[HH + c];
  yt[((long long)(b * HH + c)) * TT + t] = a * (1.f / (1.f + __expf(-gv)));
}

__global__ void dwconv1d_k(const float* __restrict__ yt, const float* __restrict__ w,
                           float* __restrict__ yc, long long n) {
  long long i = (long long)blockIdx.x * blockDim.x + threadIdx.x;
  if (i >= n) return;                 // layout (b,c,t)
  int t = (int)(i % TT);
  long long bc = i / TT;
  int c = (int)(bc % HH);
  const float* base = yt + bc * TT;
  const float* wc = w + (long long)c * K9;
  float acc = 0.f;
#pragma unroll
  for (int j = 0; j < K9; ++j) {
    int tt = t + j - K9 / 2;
    if (tt >= 0 && tt < TT) acc += base[tt] * wc[j];
  }
  yc[i] = acc;
}

__global__ void bn_silu_t_k(const float* __restrict__ yc, const float* __restrict__ g,
                            const float* __restrict__ b2, const float* __restrict__ mn,
                            const float* __restrict__ vr, bf16* __restrict__ out, long long n) {
  long long i = (long long)blockIdx.x * blockDim.x + threadIdx.x;
  if (i >= n) return;                 // in (b,c,t) -> out (b,t,c)
  int t = (int)(i % TT);
  long long bc = i / TT;
  int c = (int)(bc % HH);
  int b = (int)(bc / HH);
  float v = (yc[i] - mn[c]) * rsqrtf(vr[c] + EPSV) * g[c] + b2[c];
  v = v / (1.f + __expf(-v));
  out[((long long)(b * TT + t)) * HH + c] = (bf16)v;
}

// ---------------------------------------------------------------- subsample convs
__global__ void conv0_k(const float* __restrict__ in, const float* __restrict__ w,
                        const float* __restrict__ bias, float* __restrict__ out, long long n) {
  long long i = (long long)blockIdx.x * blockDim.x + threadIdx.x;
  if (i >= n) return;                 // out (B,CC,2048,64), 3x3 s2 p1, relu
  int x = (int)(i % 64); long long r = i / 64;
  int y = (int)(r % 2048); r /= 2048;
  int c = (int)(r % CCC);  int b = (int)(r / CCC);
  float acc = bias[c];
  for (int dy = 0; dy < 3; ++dy) {
    int iy = 2 * y + dy - 1; if (iy < 0 || iy >= 4096) continue;
    for (int dx = 0; dx < 3; ++dx) {
      int ix = 2 * x + dx - 1; if (ix < 0 || ix >= 128) continue;
      acc += in[((long long)b * 4096 + iy) * 128 + ix] * w[(c * 3 + dy) * 3 + dx];
    }
  }
  out[i] = fmaxf(acc, 0.f);
}

__global__ void dw2d_k(const float* __restrict__ in, const float* __restrict__ w,
                       const float* __restrict__ bias, float* __restrict__ out,
                       int Ti, int Fi, long long n) {
  long long i = (long long)blockIdx.x * blockDim.x + threadIdx.x;
  if (i >= n) return;                 // depthwise 3x3 s2 p1
  int To = Ti >> 1, Fo = Fi >> 1;
  int x = (int)(i % Fo); long long r = i / Fo;
  int y = (int)(r % To); r /= To;
  int c = (int)(r % CCC); int b = (int)(r / CCC);
  float acc = bias[c];
  for (int dy = 0; dy < 3; ++dy) {
    int iy = 2 * y + dy - 1; if (iy < 0 || iy >= Ti) continue;
    for (int dx = 0; dx < 3; ++dx) {
      int ix = 2 * x + dx - 1; if (ix < 0 || ix >= Fi) continue;
      acc += in[(((long long)b * CCC + c) * Ti + iy) * Fi + ix] * w[(c * 3 + dy) * 3 + dx];
    }
  }
  out[i] = acc;
}

__global__ void pw_k(const float* __restrict__ in, const float* __restrict__ w,
                     const float* __restrict__ bias, float* __restrict__ out,
                     int T, int F, long long n) {
  long long i = (long long)blockIdx.x * blockDim.x + threadIdx.x;
  if (i >= n) return;                 // 1x1 conv CC->CC, relu
  int x = (int)(i % F); long long r = i / F;
  int y = (int)(r % T); r /= T;
  int co = (int)(r % CCC); int b = (int)(r / CCC);
  float acc = bias[co];
  const float* wr = w + (long long)co * CCC;
  long long ibase = (long long)b * CCC * T * F + (long long)y * F + x;
  for (int ci = 0; ci < CCC; ++ci) acc += in[ibase + (long long)ci * T * F] * wr[ci];
  out[i] = fmaxf(acc, 0.f);
}

__global__ void reshape_sub_k(const float* __restrict__ in, bf16* __restrict__ out, long long n) {
  long long i = (long long)blockIdx.x * blockDim.x + threadIdx.x;
  if (i >= n) return;                 // (B,CC,512,16) -> (B*512, CC*16)
  int f = (int)(i % 16); long long r = i / 16;
  int c = (int)(r % CCC); r /= CCC;
  int t = (int)(r % TT); int b = (int)(r / TT);
  out[i] = (bf16)in[(((long long)b * CCC + c) * TT + t) * 16 + f];
}

// ---------------------------------------------------------------- host side
static inline unsigned gblk(long long n) { return (unsigned)((n + 255) / 256); }

static void gemm(hipStream_t st, const bf16* A, const bf16* Bm, float* Cf, bf16* Cb,
                 const float* bias, const float* add, int M, int N, int K,
                 int lda, int ldb, int ldc,
                 long long sAb, long long sAh, long long sBb, long long sBh,
                 long long sCb, long long sCh, int nB, int nH,
                 float alpha, int act, int transB)
{
  dim3 gr((N + 127) / 128, (M + 63) / 64, nB * nH);
  dim3 bl(256);
  if (transB)
    wmma_gemm<1><<<gr, bl, 0, st>>>(A, Bm, Cf, Cb, bias, add, M, N, K, lda, ldb, ldc,
                                    sAb, sAh, sBb, sBh, sCb, sCh, nH, alpha, act);
  else
    wmma_gemm<0><<<gr, bl, 0, st>>>(A, Bm, Cf, Cb, bias, add, M, N, K, lda, ldb, ldc,
                                    sAb, sAh, sBb, sBh, sCb, sCh, nH, alpha, act);
}

extern "C" void kernel_launch(void* const* d_in, const int* in_sizes, int n_in,
                              void* d_out, int out_size, void* d_ws, size_t ws_size,
                              hipStream_t stream)
{
  (void)in_sizes; (void)n_in; (void)out_size; (void)ws_size;
  const float* in_feat = (const float*)d_in[0];
#define PF(i) ((const float*)d_in[(i)])

  char* cur = (char*)d_ws;
  auto alloc = [&](size_t bytes) -> char* {
    char* p = cur; cur += (bytes + 255) & ~(size_t)255; return p;
  };

  // persistent
  float* x     = (float*)alloc((size_t)BT * HH * 4);
  bf16*  xb    = (bf16*) alloc((size_t)BT * HH * 2);
  bf16*  wbuf  = (bf16*) alloc((size_t)II * HH * 2);
  bf16*  pe_bf = (bf16*) alloc((size_t)LRELN * HH * 2);
  char*  mark  = cur;

  // ---- subsampling (region unioned with layer temporaries) ----
  {
    float* sA   = (float*)alloc((size_t)BB * CCC * 2048 * 64 * 4);
    float* sB   = (float*)alloc((size_t)BB * CCC * 1024 * 32 * 4);
    bf16*  suba = (bf16*) alloc((size_t)BT * 4096 * 2);

    long long n0 = (long long)BB * CCC * 2048 * 64;
    conv0_k<<<gblk(n0), 256, 0, stream>>>(in_feat, PF(1), PF(2), sA, n0);
    long long n1 = (long long)BB * CCC * 1024 * 32;
    dw2d_k<<<gblk(n1), 256, 0, stream>>>(sA, PF(3), PF(4), sB, 2048, 64, n1);
    pw_k  <<<gblk(n1), 256, 0, stream>>>(sB, PF(5), PF(6), sA, 1024, 32, n1);
    long long n2 = (long long)BB * CCC * 512 * 16;
    dw2d_k<<<gblk(n2), 256, 0, stream>>>(sA, PF(7), PF(8), sB, 1024, 32, n2);
    pw_k  <<<gblk(n2), 256, 0, stream>>>(sB, PF(9), PF(10), sA, 512, 16, n2);
    reshape_sub_k<<<gblk((long long)BT * 4096), 256, 0, stream>>>(sA, suba, (long long)BT * 4096);
    cast_f2b<<<gblk((long long)HH * 4096), 256, 0, stream>>>(PF(11), wbuf, (long long)HH * 4096);
    gemm(stream, suba, wbuf, x, nullptr, PF(12), nullptr, BT, HH, 4096,
         4096, 4096, HH, 0,0,0,0,0,0, 1, 1, 1.f, 0, 1);
    pe_make_k<<<gblk((long long)LRELN * HH), 256, 0, stream>>>(pe_bf);
  }

  // ---- layer temporaries (reuse subsample region) ----
  cur = mark;
  bf16*  hid  = (bf16*) alloc((size_t)BT * II * 2);
  float* qf   = (float*)alloc((size_t)BT * HH * 4);
  bf16*  kb   = (bf16*) alloc((size_t)BT * HH * 2);
  bf16*  vb   = (bf16*) alloc((size_t)BT * HH * 2);
  bf16*  qub  = (bf16*) alloc((size_t)BT * HH * 2);
  bf16*  qvb  = (bf16*) alloc((size_t)BT * HH * 2);
  bf16*  Rb   = (bf16*) alloc((size_t)LRELN * HH * 2);
  float* scc  = (float*)alloc((size_t)BB * NHH * TT * TT * 4);
  float* scp  = (float*)alloc((size_t)BB * NHH * TT * LRELN * 4);
  bf16*  Pb   = (bf16*) alloc((size_t)BB * NHH * TT * TT * 2);
  bf16*  attb = (bf16*) alloc((size_t)BT * HH * 2);
  float* y2   = (float*)alloc((size_t)BT * 2 * HH * 4);
  float* yt   = (float*)alloc((size_t)BB * HH * TT * 4);
  float* yc   = (float*)alloc((size_t)BB * HH * TT * 4);
  bf16*  ybf  = (bf16*) alloc((size_t)BT * HH * 2);

  const long long nBTH = (long long)BT * HH;

  for (int i = 0; i < NL; ++i) {
    const float* ln1g = PF(13) + (long long)i * HH;  const float* ln1b = PF(14) + (long long)i * HH;
    const float* f1w1 = PF(15) + (long long)i * II * HH;
    const float* f1w2 = PF(16) + (long long)i * HH * II;
    const float* lnag = PF(17) + (long long)i * HH;  const float* lnab = PF(18) + (long long)i * HH;
    const float* wq   = PF(19) + (long long)i * HH * HH;
    const float* wk   = PF(20) + (long long)i * HH * HH;
    const float* wv   = PF(21) + (long long)i * HH * HH;
    const float* wo   = PF(22) + (long long)i * HH * HH;
    const float* wr   = PF(23) + (long long)i * HH * HH;
    const float* bu   = PF(24) + (long long)i * HH;
    const float* bv   = PF(25) + (long long)i * HH;
    const float* lncg = PF(26) + (long long)i * HH;  const float* lncb = PF(27) + (long long)i * HH;
    const float* pw1  = PF(28) + (long long)i * 2 * HH * HH;
    const float* dwp  = PF(29) + (long long)i * HH * K9;
    const float* bng  = PF(30) + (long long)i * HH;  const float* bnb  = PF(31) + (long long)i * HH;
    const float* bnm  = PF(32) + (long long)i * HH;  const float* bnv  = PF(33) + (long long)i * HH;
    const float* pw2  = PF(34) + (long long)i * HH * HH;
    const float* ln2g = PF(35) + (long long)i * HH;  const float* ln2b = PF(36) + (long long)i * HH;
    const float* f2w1 = PF(37) + (long long)i * II * HH;
    const float* f2w2 = PF(38) + (long long)i * HH * II;
    const float* lnog = PF(39) + (long long)i * HH;  const float* lnob = PF(40) + (long long)i * HH;

    // ---- FFN 1:  x += 0.5 * silu(ln(x)@w1^T)@w2^T
    layernorm_k<<<BT, 256, 0, stream>>>(x, ln1g, ln1b, nullptr, xb);
    cast_f2b<<<gblk((long long)II * HH), 256, 0, stream>>>(f1w1, wbuf, (long long)II * HH);
    gemm(stream, xb, wbuf, nullptr, hid, nullptr, nullptr, BT, II, HH,
         HH, HH, II, 0,0,0,0,0,0, 1, 1, 1.f, 1, 1);
    cast_f2b<<<gblk((long long)HH * II), 256, 0, stream>>>(f1w2, wbuf, (long long)HH * II);
    gemm(stream, hid, wbuf, x, nullptr, nullptr, x, BT, HH, II,
         II, II, HH, 0,0,0,0,0,0, 1, 1, 0.5f, 0, 1);

    // ---- Attention
    layernorm_k<<<BT, 256, 0, stream>>>(x, lnag, lnab, nullptr, xb);
    cast_f2b<<<gblk((long long)HH * HH), 256, 0, stream>>>(wq, wbuf, (long long)HH * HH);
    gemm(stream, xb, wbuf, qf, nullptr, nullptr, nullptr, BT, HH, HH,
         HH, HH, HH, 0,0,0,0,0,0, 1, 1, 1.f, 0, 1);
    cast_f2b<<<gblk((long long)HH * HH), 256, 0, stream>>>(wk, wbuf, (long long)HH * HH);
    gemm(stream, xb, wbuf, nullptr, kb, nullptr, nullptr, BT, HH, HH,
         HH, HH, HH, 0,0,0,0,0,0, 1, 1, 1.f, 0, 1);
    cast_f2b<<<gblk((long long)HH * HH), 256, 0, stream>>>(wv, wbuf, (long long)HH * HH);
    gemm(stream, xb, wbuf, nullptr, vb, nullptr, nullptr, BT, HH, HH,
         HH, HH, HH, 0,0,0,0,0,0, 1, 1, 1.f, 0, 1);
    q_bias_k<<<gblk(nBTH), 256, 0, stream>>>(qf, bu, bv, qub, qvb, nBTH);
    cast_f2b<<<gblk((long long)HH * HH), 256, 0, stream>>>(wr, wbuf, (long long)HH * HH);
    gemm(stream, pe_bf, wbuf, nullptr, Rb, nullptr, nullptr, LRELN, HH, HH,
         HH, HH, HH, 0,0,0,0,0,0, 1, 1, 1.f, 0, 1);
    // content scores: (q+bu) @ k^T  per (b,h)
    gemm(stream, qub, kb, scc, nullptr, nullptr, nullptr, TT, TT, HDD,
         HH, HH, TT,
         (long long)TT * HH, HDD, (long long)TT * HH, HDD,
         (long long)NHH * TT * TT, (long long)TT * TT, BB, NHH, 1.f, 0, 1);
    // positional scores: (q+bv) @ rel_k^T
    gemm(stream, qvb, Rb, scp, nullptr, nullptr, nullptr, TT, LRELN, HDD,
         HH, HH, LRELN,
         (long long)TT * HH, HDD, 0, HDD,
         (long long)NHH * TT * LRELN, (long long)TT * LRELN, BB, NHH, 1.f, 0, 1);
    softmax_rel_k<<<dim3(TT, BB * NHH), 256, 0, stream>>>(scc, scp, Pb);
    // context: P @ v  (B is K x N => transB=0), write directly interleaved (b,t,h,d)
    gemm(stream, Pb, vb, nullptr, attb, nullptr, nullptr, TT, HDD, TT,
         TT, HH, HH,
         (long long)NHH * TT * TT, (long long)TT * TT, (long long)TT * HH, HDD,
         (long long)TT * HH, HDD, BB, NHH, 1.f, 0, 0);
    cast_f2b<<<gblk((long long)HH * HH), 256, 0, stream>>>(wo, wbuf, (long long)HH * HH);
    gemm(stream, attb, wbuf, x, nullptr, nullptr, x, BT, HH, HH,
         HH, HH, HH, 0,0,0,0,0,0, 1, 1, 1.f, 0, 1);

    // ---- Conv module
    layernorm_k<<<BT, 256, 0, stream>>>(x, lncg, lncb, nullptr, xb);
    cast_f2b<<<gblk((long long)2 * HH * HH), 256, 0, stream>>>(pw1, wbuf, (long long)2 * HH * HH);
    gemm(stream, xb, wbuf, y2, nullptr, nullptr, nullptr, BT, 2 * HH, HH,
         HH, HH, 2 * HH, 0,0,0,0,0,0, 1, 1, 1.f, 0, 1);
    glu_t_k   <<<gblk(nBTH), 256, 0, stream>>>(y2, yt, nBTH);
    dwconv1d_k<<<gblk(nBTH), 256, 0, stream>>>(yt, dwp, yc, nBTH);
    bn_silu_t_k<<<gblk(nBTH), 256, 0, stream>>>(yc, bng, bnb, bnm, bnv, ybf, nBTH);
    cast_f2b<<<gblk((long long)HH * HH), 256, 0, stream>>>(pw2, wbuf, (long long)HH * HH);
    gemm(stream, ybf, wbuf, x, nullptr, nullptr, x, BT, HH, HH,
         HH, HH, HH, 0,0,0,0,0,0, 1, 1, 1.f, 0, 1);

    // ---- FFN 2
    layernorm_k<<<BT, 256, 0, stream>>>(x, ln2g, ln2b, nullptr, xb);
    cast_f2b<<<gblk((long long)II * HH), 256, 0, stream>>>(f2w1, wbuf, (long long)II * HH);
    gemm(stream, xb, wbuf, nullptr, hid, nullptr, nullptr, BT, II, HH,
         HH, HH, II, 0,0,0,0,0,0, 1, 1, 1.f, 1, 1);
    cast_f2b<<<gblk((long long)HH * II), 256, 0, stream>>>(f2w2, wbuf, (long long)HH * II);
    gemm(stream, hid, wbuf, x, nullptr, nullptr, x, BT, HH, II,
         II, II, HH, 0,0,0,0,0,0, 1, 1, 0.5f, 0, 1);

    // ---- output layernorm (last layer writes straight to d_out)
    float* outp = (i == NL - 1) ? (float*)d_out : x;
    layernorm_k<<<BT, 256, 0, stream>>>(x, lnog, lnob, outp, nullptr);
  }
#undef PF
}
